// BiLSTM_CRF_83786222010410
// MI455X (gfx1250) — matrix operand order repaired and checked by reading, not compiled, not run on hardware
//
#include <hip/hip_runtime.h>
#include <hip/hip_bf16.h>
#include <math.h>

typedef __attribute__((ext_vector_type(16))) _Float16 v16h;
typedef __attribute__((ext_vector_type(8)))  _Float16 v8h;
typedef __attribute__((ext_vector_type(8)))  float    v8f;
typedef __attribute__((ext_vector_type(4)))  int      v4i;

#define SLEN 4096
#define EDIM 256
#define HH   256
#define GATES 1024   /* 4*HH */
#define HDIM 512
#define TTAGS 9
#define TSTART 7
#define TSTOP 8
#define NEGV (-10000.0f)

#if __has_builtin(__builtin_amdgcn_global_load_async_to_lds_b128) && \
    __has_builtin(__builtin_amdgcn_s_wait_asynccnt)
#define HAVE_ASYNC_LDS 1
#else
#define HAVE_ASYNC_LDS 0
#endif

#if HAVE_ASYNC_LDS
typedef __attribute__((address_space(1))) void as1_void;
typedef __attribute__((address_space(3))) void as3_void;
typedef __attribute__((address_space(1))) v4i* as1_v4i_p;
typedef __attribute__((address_space(3))) v4i* as3_v4i_p;
__device__ __forceinline__ void async_g2l_b128(const void* g, void* l) {
  // param0: v4i addrspace(1)* (global src), param1: v4i addrspace(3)* (LDS dst), offset, cpol
  __builtin_amdgcn_global_load_async_to_lds_b128(
      (as1_v4i_p)(as1_void*)(void*)g, (as3_v4i_p)(as3_void*)l, 0, 0);
}
#endif

__device__ __forceinline__ float sigmoidf_(float x) { return 1.0f / (1.0f + __expf(-x)); }
__device__ __forceinline__ float tanhf_(float x)    { return 1.0f - 2.0f / (__expf(2.0f * x) + 1.0f); }

// ---------------- prep kernels ----------------
__global__ void gather_embed_k(const int* __restrict__ sent,
                               const float* __restrict__ embed,
                               _Float16* __restrict__ Xh) {
  int s = blockIdx.x, e = threadIdx.x;              // 4096 x 256
  Xh[s * EDIM + e] = (_Float16)embed[(long)sent[s] * EDIM + e];
}

__global__ void f32_to_f16_k(const float* __restrict__ src, _Float16* __restrict__ dst, int n) {
  int i = blockIdx.x * 256 + threadIdx.x;
  if (i < n) dst[i] = (_Float16)src[i];
}

__global__ void pad_wout_k(const float* __restrict__ Wout, _Float16* __restrict__ dst) {
  int i = blockIdx.x * 256 + threadIdx.x;           // 16*512
  int r = i >> 9, c = i & 511;
  dst[i] = (r < TTAGS) ? (_Float16)Wout[r * HDIM + c] : (_Float16)0.0f;
}

// ---------------- input-gate GEMM: Xg[4096,1024] = Xh @ Wih^T + b_ih + b_hh ----------------
// block = 256 threads (8 waves); grid = (256 m-tiles, 4 n-groups);
// each wave owns two adjacent 16x16 tiles -> A fragment reused, two independent WMMA chains.
__global__ void gemm_xg_k(const _Float16* __restrict__ Xh,     // [S, E]
                          const _Float16* __restrict__ Wh,     // [1024, E] row-major
                          const float* __restrict__ b_ih,
                          const float* __restrict__ b_hh,
                          float* __restrict__ Xg) {             // [S, 1024]
  int wave = threadIdx.x >> 5, lane = threadIdx.x & 31;
  int mtile = blockIdx.x;
  int nbase = (blockIdx.y * 8 + wave) * 2;          // two n-tiles per wave
  int m = lane & 15, hi = lane >> 4;
  v8f acc0 = {}, acc1 = {};
  const _Float16* arow  = Xh + (long)(mtile * 16 + m) * EDIM;
  const _Float16* brow0 = Wh + (long)((nbase + 0) * 16 + m) * EDIM;
  const _Float16* brow1 = Wh + (long)((nbase + 1) * 16 + m) * EDIM;
  for (int k0 = 0; k0 < EDIM; k0 += 32) {
    v8h alo = *(const v8h*)(arow + k0 + hi * 16);
    v8h ahi = *(const v8h*)(arow + k0 + hi * 16 + 8);
    v16h av;
#pragma unroll
    for (int e = 0; e < 8; ++e) { av[e] = alo[e]; av[8 + e] = ahi[e]; }
    v16h bv0 = *(const v16h*)(brow0 + k0 + hi * 16);
    v16h bv1 = *(const v16h*)(brow1 + k0 + hi * 16);
    acc0 = __builtin_amdgcn_wmma_f32_16x16x32_f16(false, av, false, bv0, (short)0, acc0, false, false);
    acc1 = __builtin_amdgcn_wmma_f32_16x16x32_f16(false, av, false, bv1, (short)0, acc1, false, false);
  }
#pragma unroll
  for (int r = 0; r < 8; ++r) {
    int row = mtile * 16 + hi * 8 + r;
    int c0 = (nbase + 0) * 16 + m;
    int c1 = (nbase + 1) * 16 + m;
    Xg[(long)row * GATES + c0] = acc0[r] + b_ih[c0] + b_hh[c0];
    Xg[(long)row * GATES + c1] = acc1[r] + b_ih[c1] + b_hh[c1];
  }
}

// ---------------- sequential bidirectional LSTM ----------------
// grid = 2 (dir), block = 1024 (32 waves). Each step: g = Xg[t] + Whh @ h via WMMA
// (h padded to M=16, only row 0 live). h and g exchanged through LDS.
// Xg row for step t+1 is staged into LDS with async load-to-LDS, double buffered.
__global__ void lstm_seq_k(const float* __restrict__ XgF, const float* __restrict__ XgB,
                           const _Float16* __restrict__ WhF, const _Float16* __restrict__ WhB,
                           const float* __restrict__ h0, const float* __restrict__ c0,
                           _Float16* __restrict__ Yh) {          // [S, 512] f16
  const int dir = blockIdx.x;
  const float* Xg = dir ? XgB : XgF;
  const _Float16* W = dir ? WhB : WhF;
  __shared__ _Float16 h_sh[HH];
  __shared__ float g_sh[GATES];
#if HAVE_ASYNC_LDS
  __shared__ float xg_buf[2][GATES];
#endif
  int tid = threadIdx.x;
  int wave = tid >> 5, lane = tid & 31;
  int m = lane & 15, hi = lane >> 4;
  if (tid < HH) h_sh[tid] = (_Float16)h0[dir * HH + tid];
  float c = (tid < HH) ? c0[dir * HH + tid] : 0.0f;
#if HAVE_ASYNC_LDS
  {
    int te0 = dir ? (SLEN - 1) : 0;
    if (tid < 256) async_g2l_b128(Xg + (long)te0 * GATES + tid * 4, &xg_buf[0][tid * 4]);
    __builtin_amdgcn_s_wait_asynccnt(0);
  }
#endif
  __syncthreads();

  const int nbase = wave * 2;                 // 64 n-tiles over 32 waves
  for (int t = 0; t < SLEN; ++t) {
    int te = dir ? (SLEN - 1 - t) : t;
#if HAVE_ASYNC_LDS
    if (t + 1 < SLEN && tid < 256) {          // stage next step's gates while WMMA runs
      int tn = dir ? (te - 1) : (te + 1);
      async_g2l_b128(Xg + (long)tn * GATES + tid * 4, &xg_buf[(t + 1) & 1][tid * 4]);
    }
#endif
    // two independent WMMA accumulation chains, A fragment built once per k-block
    v8f acc0 = {}, acc1 = {};
    for (int k0 = 0; k0 < HH; k0 += 32) {
      v16h av;
#pragma unroll
      for (int e = 0; e < 8; ++e) {
        _Float16 lo = h_sh[k0 + hi * 16 + e];
        _Float16 hv = h_sh[k0 + hi * 16 + 8 + e];
        av[e]     = (m == 0) ? lo : (_Float16)0.0f;
        av[8 + e] = (m == 0) ? hv : (_Float16)0.0f;
      }
      v16h bv0 = *(const v16h*)(W + (long)((nbase + 0) * 16 + m) * HH + k0 + hi * 16);
      v16h bv1 = *(const v16h*)(W + (long)((nbase + 1) * 16 + m) * HH + k0 + hi * 16);
      acc0 = __builtin_amdgcn_wmma_f32_16x16x32_f16(false, av, false, bv0, (short)0, acc0, false, false);
      acc1 = __builtin_amdgcn_wmma_f32_16x16x32_f16(false, av, false, bv1, (short)0, acc1, false, false);
    }
    if (hi == 0) {                            // D row M=0 lives in r=0, lanes 0..15
      g_sh[(nbase + 0) * 16 + m] = acc0[0];
      g_sh[(nbase + 1) * 16 + m] = acc1[0];
    }
    __syncthreads();
    float hnew = 0.0f;
    if (tid < HH) {
#if HAVE_ASYNC_LDS
      const float* xg = &xg_buf[t & 1][0];    // staged in LDS
#else
      const float* xg = Xg + (long)te * GATES;
#endif
      float gi = xg[tid]          + g_sh[tid];
      float gf = xg[HH + tid]     + g_sh[HH + tid];
      float gg = xg[2 * HH + tid] + g_sh[2 * HH + tid];
      float go = xg[3 * HH + tid] + g_sh[3 * HH + tid];
      float i_ = sigmoidf_(gi), f_ = sigmoidf_(gf), o_ = sigmoidf_(go);
      float g_ = tanhf_(gg);
      c = f_ * c + i_ * g_;
      hnew = o_ * tanhf_(c);
      Yh[(long)te * HDIM + dir * HH + tid] = (_Float16)hnew;
#if !HAVE_ASYNC_LDS
      int tn = dir ? (te - 1) : (te + 1);
      if (tn >= 0 && tn < SLEN) __builtin_prefetch(Xg + (long)tn * GATES + tid, 0, 0);
#endif
    }
    __syncthreads();
    if (tid < HH) h_sh[tid] = (_Float16)hnew;
#if HAVE_ASYNC_LDS
    __builtin_amdgcn_s_wait_asynccnt(0);      // next xg buffer resident before next step
#endif
    __syncthreads();
  }
}

// ---------------- output GEMM: Frames[4096,16] = Yh @ Wout^T (+b) ----------------
// K=512 split into two independent 8-deep WMMA chains to hide the D->C RAW hazard.
__global__ void gemm_frames_k(const _Float16* __restrict__ Yh,     // [S, 512]
                              const _Float16* __restrict__ WoutH,  // [16, 512] (rows>=9 zero)
                              const float* __restrict__ b_out,
                              float* __restrict__ Frames) {        // [S, 16]
  int wave = threadIdx.x >> 5, lane = threadIdx.x & 31;
  int mtile = blockIdx.x * 8 + wave;    // 256 tiles with grid 32
  int m = lane & 15, hi = lane >> 4;
  v8f acc0 = {}, acc1 = {};
  const _Float16* arow = Yh + (long)(mtile * 16 + m) * HDIM;
  const _Float16* brow = WoutH + (long)m * HDIM;
  for (int k0 = 0; k0 < HDIM / 2; k0 += 32) {
    int kA = k0, kB = k0 + HDIM / 2;
    v8h alo0 = *(const v8h*)(arow + kA + hi * 16);
    v8h ahi0 = *(const v8h*)(arow + kA + hi * 16 + 8);
    v8h alo1 = *(const v8h*)(arow + kB + hi * 16);
    v8h ahi1 = *(const v8h*)(arow + kB + hi * 16 + 8);
    v16h av0, av1;
#pragma unroll
    for (int e = 0; e < 8; ++e) {
      av0[e] = alo0[e]; av0[8 + e] = ahi0[e];
      av1[e] = alo1[e]; av1[8 + e] = ahi1[e];
    }
    v16h bv0 = *(const v16h*)(brow + kA + hi * 16);
    v16h bv1 = *(const v16h*)(brow + kB + hi * 16);
    acc0 = __builtin_amdgcn_wmma_f32_16x16x32_f16(false, av0, false, bv0, (short)0, acc0, false, false);
    acc1 = __builtin_amdgcn_wmma_f32_16x16x32_f16(false, av1, false, bv1, (short)0, acc1, false, false);
  }
  v8f acc = acc0 + acc1;
#pragma unroll
  for (int r = 0; r < 8; ++r) {
    int row = mtile * 16 + hi * 8 + r;
    Frames[row * 16 + m] = acc[r] + ((m < TTAGS) ? b_out[m] : 0.0f);
  }
}

// ---------------- Viterbi/CRF: forward logsumexp + argmax backpointers + backtrack ----------------
// single workgroup, single wave; backpointer table kept in LDS (36 KB) so the
// 4095-step backtrack walks LDS, not HBM.
__global__ void viterbi_k(const float* __restrict__ Frames,  // [S, 16]
                          const float* __restrict__ trans,   // [9, 9]
                          float* __restrict__ out) {         // [1 + S]
  __shared__ unsigned char bp_sh[SLEN][TTAGS];
  __shared__ float alpha_sh[16];
  int lane = threadIdx.x;
  float tr[TTAGS];
  if (lane < TTAGS) {
#pragma unroll
    for (int i = 0; i < TTAGS; ++i) tr[i] = trans[i * TTAGS + lane];  // column `lane`
    alpha_sh[lane] = (lane == TSTART) ? 0.0f : NEGV;
  }
  __syncthreads();
  for (int t = 0; t < SLEN; ++t) {
    float a[TTAGS];
#pragma unroll
    for (int i = 0; i < TTAGS; ++i) a[i] = alpha_sh[i];
    __syncthreads();
    if (lane < TTAGS) {
      float fr = Frames[t * 16 + lane];
      float s[TTAGS];
      float mx = -3.0e38f; int bi = 0;
#pragma unroll
      for (int i = 0; i < TTAGS; ++i) {
        s[i] = a[i] + fr + tr[i];
        if (s[i] > mx) { mx = s[i]; bi = i; }
      }
      float sum = 0.0f;
#pragma unroll
      for (int i = 0; i < TTAGS; ++i) sum += __expf(s[i] - mx);
      alpha_sh[lane] = mx + __logf(sum);
      bp_sh[t][lane] = (unsigned char)bi;
    }
    __syncthreads();
  }
  if (lane == 0) {
    float mx = -3.0e38f; int best = 0;
    float fin[TTAGS];
    for (int j = 0; j < TTAGS; ++j) {
      fin[j] = alpha_sh[j] + trans[j * TTAGS + TSTOP];
      if (fin[j] > mx) { mx = fin[j]; best = j; }
    }
    float sum = 0.0f;
    for (int j = 0; j < TTAGS; ++j) sum += __expf(fin[j] - mx);
    out[0] = mx + __logf(sum);
    int tag = best;
    out[1 + SLEN - 1] = (float)tag;
    for (int t = SLEN - 1; t >= 1; --t) {
      tag = bp_sh[t][tag];
      out[t] = (float)tag;        // out[1 + (t-1)]
    }
  }
}

// ---------------- host side ----------------
extern "C" void kernel_launch(void* const* d_in, const int* in_sizes, int n_in,
                              void* d_out, int out_size, void* d_ws, size_t ws_size,
                              hipStream_t stream) {
  const int*   sentence = (const int*)  d_in[0];
  const float* embed    = (const float*)d_in[1];
  const float* w_ih_f   = (const float*)d_in[2];
  const float* w_hh_f   = (const float*)d_in[3];
  const float* b_ih_f   = (const float*)d_in[4];
  const float* b_hh_f   = (const float*)d_in[5];
  const float* w_ih_b   = (const float*)d_in[6];
  const float* w_hh_b   = (const float*)d_in[7];
  const float* b_ih_b   = (const float*)d_in[8];
  const float* b_hh_b   = (const float*)d_in[9];
  const float* h0       = (const float*)d_in[10];
  const float* c0       = (const float*)d_in[11];
  const float* W_out    = (const float*)d_in[12];
  const float* b_out    = (const float*)d_in[13];
  const float* trans    = (const float*)d_in[14];
  float* out = (float*)d_out;

  char* ws = (char*)d_ws;
  size_t off = 0;
  auto alloc = [&](size_t bytes) { char* p = ws + off; off = (off + bytes + 255) & ~(size_t)255; return p; };
  _Float16* Xh     = (_Float16*)alloc((size_t)SLEN * EDIM * 2);
  _Float16* WihF_h = (_Float16*)alloc((size_t)GATES * EDIM * 2);
  _Float16* WihB_h = (_Float16*)alloc((size_t)GATES * EDIM * 2);
  _Float16* WhhF_h = (_Float16*)alloc((size_t)GATES * HH * 2);
  _Float16* WhhB_h = (_Float16*)alloc((size_t)GATES * HH * 2);
  _Float16* WoutH  = (_Float16*)alloc((size_t)16 * HDIM * 2);
  float*    XgF    = (float*)   alloc((size_t)SLEN * GATES * 4);
  float*    XgB    = (float*)   alloc((size_t)SLEN * GATES * 4);
  _Float16* Yh     = (_Float16*)alloc((size_t)SLEN * HDIM * 2);
  float*    Frames = (float*)   alloc((size_t)SLEN * 16 * 4);
  (void)ws_size; (void)n_in; (void)in_sizes; (void)out_size;

  // prep: gather embedding + f16 weight copies
  gather_embed_k<<<SLEN, EDIM, 0, stream>>>(sentence, embed, Xh);
  f32_to_f16_k<<<(GATES * EDIM + 255) / 256, 256, 0, stream>>>(w_ih_f, WihF_h, GATES * EDIM);
  f32_to_f16_k<<<(GATES * EDIM + 255) / 256, 256, 0, stream>>>(w_ih_b, WihB_h, GATES * EDIM);
  f32_to_f16_k<<<(GATES * HH + 255) / 256, 256, 0, stream>>>(w_hh_f, WhhF_h, GATES * HH);
  f32_to_f16_k<<<(GATES * HH + 255) / 256, 256, 0, stream>>>(w_hh_b, WhhB_h, GATES * HH);
  pad_wout_k<<<(16 * HDIM) / 256, 256, 0, stream>>>(W_out, WoutH);

  // big parallel GEMMs (WMMA, two n-tiles per wave)
  gemm_xg_k<<<dim3(SLEN / 16, GATES / 256), 256, 0, stream>>>(Xh, WihF_h, b_ih_f, b_hh_f, XgF);
  gemm_xg_k<<<dim3(SLEN / 16, GATES / 256), 256, 0, stream>>>(Xh, WihB_h, b_ih_b, b_hh_b, XgB);

  // sequential recurrence, one WGP per direction (WMMA matvec per step, async-LDS staged gates)
  lstm_seq_k<<<2, 1024, 0, stream>>>(XgF, XgB, WhhF_h, WhhB_h, h0, c0, Yh);

  // emissions (WMMA, N padded to 16)
  gemm_frames_k<<<SLEN / 128, 256, 0, stream>>>(Yh, WoutH, b_out, Frames);

  // CRF forward + backtrack
  viterbi_k<<<1, 32, 0, stream>>>(Frames, trans, out);
}
